// DecoderRNN_89361089561133
// MI455X (gfx1250) — matrix operand order repaired and checked by reading, not compiled, not run on hardware
//
#include <hip/hip_runtime.h>
#include <math.h>
#include <stdint.h>

typedef _Float16 half_t;
typedef __attribute__((ext_vector_type(16))) _Float16 v16h;
typedef __attribute__((ext_vector_type(8)))  _Float16 v8h;
typedef __attribute__((ext_vector_type(8)))  float    v8f;
typedef __attribute__((ext_vector_type(4)))  uint32_t u32x4;
typedef __attribute__((ext_vector_type(8)))  int      i32x8;
typedef __attribute__((ext_vector_type(4)))  int      i32x4;

#define VOCAB  10000
#define HID    1024
#define WORD   512
#define TSTEPS 27            // MAX_LEN - 1
#define MAXLEN 28
#define BATCH  256
#define LENC   36

// GEMM tiling
#define KB        64                       // K halves staged per slab
#define BROWS     256                      // N rows staged per slab
#define ROW_BYTES 144                      // 128 data + 16 pad (bank-conflict-free)
#define SLAB_BYTES (BROWS * ROW_BYTES)     // 36864

#if __has_builtin(__builtin_amdgcn_tensor_load_to_lds)
#define USE_TDM 1
#else
#define USE_TDM 0
#endif

union Frag16 { v16h v; v8h h[2]; };

#if USE_TDM
// Issue one TDM load of a B slab: rows [n0, n0+256) x K-halves [kb, kb+64)
// of W (row-major, stride ldw halves) into LDS at lds_addr, padded rows.
__device__ __forceinline__ void tdm_load_b_slab(uint32_t lds_addr,
                                                const half_t* gsrc,
                                                uint32_t remN, uint32_t remK,
                                                uint32_t ldw) {
    uint64_t ga = (uint64_t)(uintptr_t)gsrc;
    u32x4 g0;
    g0[0] = 1u;                                   // count=1, user mode
    g0[1] = lds_addr;                             // LDS byte address
    g0[2] = (uint32_t)ga;                         // global_addr[31:0]
    g0[3] = ((uint32_t)(ga >> 32) & 0x01FFFFFFu)  // global_addr[56:32]
            | 0x80000000u;                        // type=2 ("image")
    i32x8 g1;
    // data_size=1 (2B), pad_enable=1, pad_interval=4 (128B), pad_amount=3 (16B)
    g1[0] = (int)((1u << 16) | (1u << 20) | (4u << 22) | (3u << 25));
    g1[1] = (int)((remK & 0xFFFFu) << 16);                    // tensor_dim0[15:0]
    g1[2] = (int)((remK >> 16) | ((remN & 0xFFFFu) << 16));   // td0[31:16] | td1[15:0]
    g1[3] = (int)((remN >> 16) | ((uint32_t)KB << 16));       // td1[31:16] | tile_dim0
    g1[4] = (int)BROWS;                                       // tile_dim1 (tile_dim2=0)
    g1[5] = (int)ldw;                                         // tensor_dim0_stride[31:0]
    g1[6] = 0;
    g1[7] = 0;
    i32x4 gz = {0, 0, 0, 0};
#if __clang_major__ >= 23
    i32x8 gz8 = {0, 0, 0, 0, 0, 0, 0, 0};
    __builtin_amdgcn_tensor_load_to_lds(g0, g1, gz, gz, gz8, 0);
#else
    __builtin_amdgcn_tensor_load_to_lds(g0, g1, gz, gz, 0);
#endif
}
#endif

// ---------------------------------------------------------------------------
// WMMA GEMM with TDM-staged, LDS-double-buffered B:
//   C[M,N] (f32) = A[M,K] (f16 row-major) x W[N,K]^T (f16 row-major) + bias
// M multiple of 32; K multiple of 64. Block = 256 thr = 8 waves (2M x 4N),
// block tile 32(M) x 256(N); B slab 256 rows x 64 K per stage.
// ---------------------------------------------------------------------------
__global__ __launch_bounds__(256)
void gemm_f16_wmma(const half_t* __restrict__ A, int lda,
                   const half_t* __restrict__ W, int ldw,
                   float* __restrict__ C, int ldc,
                   const float* __restrict__ bias,
                   int N, int K) {
    __shared__ __attribute__((aligned(16))) char lds_b[2 * SLAB_BYTES];

    const int tid  = threadIdx.x;
    const int lane = tid & 31;
    const int wv   = tid >> 5;
    const int n16  = lane & 15;
    const int hi   = lane >> 4;

    const int m0    = blockIdx.y * 32 + (wv & 1) * 16;
    const int n0blk = blockIdx.x * BROWS;
    const int nwv   = (wv >> 1) * 64;          // wave's N offset inside slab

    const v8f zero = {0.f,0.f,0.f,0.f,0.f,0.f,0.f,0.f};
    v8f acc[4];
#pragma unroll
    for (int i = 0; i < 4; ++i) acc[i] = zero;

    const half_t* pa = A + (long)(m0 + n16) * lda + hi * 8;
    const half_t* wblk = W + (long)n0blk * ldw;
    const uint32_t lds_base = (uint32_t)(uintptr_t)&lds_b[0];

#if USE_TDM
    if (wv == 0)
        tdm_load_b_slab(lds_base, wblk, (uint32_t)(N - n0blk), (uint32_t)K,
                        (uint32_t)ldw);
#endif

    for (int kb = 0; kb < K; kb += KB) {
        const int p = (kb >> 6) & 1;
#if USE_TDM
        if (wv == 0) {
            if (kb + KB < K) {
                tdm_load_b_slab(lds_base + (uint32_t)((p ^ 1) * SLAB_BYTES),
                                wblk + kb + KB, (uint32_t)(N - n0blk),
                                (uint32_t)(K - kb - KB), (uint32_t)ldw);
                __builtin_amdgcn_s_wait_tensorcnt(1);   // current slab done
            } else {
                __builtin_amdgcn_s_wait_tensorcnt(0);
            }
        }
        __syncthreads();                                // slab p visible to all
#else
        // Fallback: cooperative synchronous copy of slab p (row per thread)
        {
            int r = tid;                                // 0..255
            int gr = n0blk + r;
            const half_t* src = wblk + (long)r * ldw + kb;
            v8h* dst = (v8h*)(lds_b + p * SLAB_BYTES + r * ROW_BYTES);
            v8h zz = {};
#pragma unroll
            for (int j = 0; j < 8; ++j)
                dst[j] = (gr < N) ? *(const v8h*)(src + j * 8) : zz;
        }
        __syncthreads();
#endif
        const char* slab = lds_b + p * SLAB_BYTES;
        if (kb + KB < K) __builtin_prefetch(pa + kb + KB, 0, 0);
#pragma unroll
        for (int kk = 0; kk < KB; kk += 32) {
            Frag16 a;
            a.h[0] = *(const v8h*)(pa + kb + kk);
            a.h[1] = *(const v8h*)(pa + kb + kk + 16);
#pragma unroll
            for (int i = 0; i < 4; ++i) {
                const char* pb = slab + (nwv + 16 * i + n16) * ROW_BYTES
                               + (kk + hi * 8) * 2;
                Frag16 b;
                b.h[0] = *(const v8h*)(pb);
                b.h[1] = *(const v8h*)(pb + 32);
                acc[i] = __builtin_amdgcn_wmma_f32_16x16x32_f16(
                    false, a.v, false, b.v, (short)0, acc[i], false, false);
            }
        }
        __syncthreads();                                // done reading slab p
    }

#pragma unroll
    for (int i = 0; i < 4; ++i) {
        int col = n0blk + nwv + 16 * i + n16;
        if (col < N) {
            float bv = (bias != nullptr) ? bias[col] : 0.f;
#pragma unroll
            for (int r = 0; r < 8; ++r) {
                int row = m0 + r + 8 * hi;
                C[(long)row * ldc + col] = acc[i][r] + bv;
            }
        }
    }
}

// ---------------------------------------------------------------------------
// One-time prep kernels
// ---------------------------------------------------------------------------
__global__ void f32_to_f16_kernel(const float* __restrict__ in,
                                  half_t* __restrict__ out, int n) {
    int i = blockIdx.x * 256 + threadIdx.x;
    if (i < n) out[i] = (half_t)in[i];
}

// attn_W (2048,1024) -> WhT[n][k] = attn_W[k][n], WeT[n][k] = attn_W[1024+k][n]
__global__ void attnw_transpose_kernel(const float* __restrict__ attnW,
                                       half_t* __restrict__ WhT,
                                       half_t* __restrict__ WeT) {
    int i = blockIdx.x * 256 + threadIdx.x;   // over 1024*1024
    int n = i >> 10;
    int k = i & 1023;
    WhT[(long)n * 1024 + k] = (half_t)attnW[(long)k * 1024 + n];
    WeT[(long)n * 1024 + k] = (half_t)attnW[(long)(1024 + k) * 1024 + n];
}

__global__ void embed_gather_kernel(const int* __restrict__ targets,
                                    const float* __restrict__ emb,
                                    half_t* __restrict__ out) {
    int i = blockIdx.x * 256 + threadIdx.x;   // BATCH*TSTEPS*WORD
    int j = i & (WORD - 1);
    int t = (i >> 9) % TSTEPS;
    int b = i / (TSTEPS * WORD);
    int tok = targets[b * MAXLEN + t];
    out[i] = (half_t)emb[(long)tok * WORD + j];
}

__global__ void h_init_kernel(const float* __restrict__ eh,
                              float* __restrict__ h,
                              half_t* __restrict__ h16) {
    int i = blockIdx.x * 256 + threadIdx.x;   // BATCH*HID
    float v = eh[i];
    h[i] = v;
    h16[i] = (half_t)v;
}

// ---------------------------------------------------------------------------
// Per-step kernels
// ---------------------------------------------------------------------------
__global__ void x_emb_copy_kernel(const half_t* __restrict__ emb,
                                  half_t* __restrict__ x, int t) {
    int i = blockIdx.x * 256 + threadIdx.x;   // BATCH*WORD
    int b = i >> 9;
    int j = i & (WORD - 1);
    x[b * (HID + WORD) + j] = emb[((long)b * TSTEPS + t) * WORD + j];
}

__global__ __launch_bounds__(256)
void attn_ctx_kernel(const float* __restrict__ Wh,
                     const float* __restrict__ encU,
                     const float* __restrict__ enc,
                     const float* __restrict__ attn_v,
                     half_t* __restrict__ x) {
    int b = blockIdx.x;
    int tid = threadIdx.x;
    int lane = tid & 31;
    int wv = tid >> 5;

    __shared__ float s[LENC];
    __shared__ float prob[LENC];
    __shared__ float red2[2];

    const float* whb = Wh + (long)b * HID;
    for (int l = wv; l < LENC; l += 8) {
        const float* eu = encU + ((long)b * LENC + l) * HID;
        float acc = 0.f;
        for (int j = lane; j < HID; j += 32)
            acc += attn_v[j] * tanhf(whb[j] + eu[j]);
        for (int o = 16; o >= 1; o >>= 1) acc += __shfl_xor(acc, o);
        if (lane == 0) s[l] = acc;
    }
    __syncthreads();
    if (tid == 0) {
        float mx = s[0];
        for (int l = 1; l < LENC; ++l) mx = fmaxf(mx, s[l]);
        float sum = 0.f;
        for (int l = 0; l < LENC; ++l) sum += expf(s[l] - mx);
        red2[0] = mx;
        red2[1] = sum;
    }
    __syncthreads();
    float mx = red2[0];
    float inv = 1.f / red2[1];
    if (tid < LENC) prob[tid] = expf(s[tid] - mx) * inv;
    __syncthreads();
    for (int j = tid; j < HID; j += 256) {
        const float* eb = enc + (long)b * LENC * HID + j;
        float acc = 0.f;
#pragma unroll
        for (int l = 0; l < LENC; ++l) acc += prob[l] * eb[(long)l * HID];
        x[b * (HID + WORD) + WORD + j] = (half_t)acc;
    }
}

__global__ void gru_update_kernel(const float* __restrict__ gi,
                                  const float* __restrict__ gh,
                                  float* __restrict__ h,
                                  half_t* __restrict__ h16) {
    int i = blockIdx.x * 256 + threadIdx.x;   // BATCH*HID
    int b = i >> 10;
    int j = i & (HID - 1);
    const float* gib = gi + (long)b * 3 * HID;
    const float* ghb = gh + (long)b * 3 * HID;
    float r = 1.f / (1.f + expf(-(gib[j] + ghb[j])));
    float z = 1.f / (1.f + expf(-(gib[HID + j] + ghb[HID + j])));
    float n = tanhf(gib[2 * HID + j] + r * ghb[2 * HID + j]);
    float hold = h[i];
    float hn = (1.f - z) * n + z * hold;
    h[i] = hn;
    h16[i] = (half_t)hn;
}

__global__ __launch_bounds__(256)
void log_softmax_kernel(const float* __restrict__ logits,
                        float* __restrict__ out, int t) {
    int b = blockIdx.x;
    int tid = threadIdx.x;
    int lane = tid & 31;
    int wv = tid >> 5;
    __shared__ float red[8];

    const float* row = logits + (long)b * VOCAB;
    float mx = -1e30f;
    for (int v = tid; v < VOCAB; v += 256) mx = fmaxf(mx, row[v]);
    for (int o = 16; o >= 1; o >>= 1) mx = fmaxf(mx, __shfl_xor(mx, o));
    if (lane == 0) red[wv] = mx;
    __syncthreads();
    float m = red[0];
#pragma unroll
    for (int i = 1; i < 8; ++i) m = fmaxf(m, red[i]);
    __syncthreads();

    float sum = 0.f;
    for (int v = tid; v < VOCAB; v += 256) sum += expf(row[v] - m);
    for (int o = 16; o >= 1; o >>= 1) sum += __shfl_xor(sum, o);
    if (lane == 0) red[wv] = sum;
    __syncthreads();
    float s = 0.f;
#pragma unroll
    for (int i = 0; i < 8; ++i) s += red[i];
    float lse = m + logf(s);

    float* orow = out + ((long)b * TSTEPS + t) * VOCAB;
    for (int v = tid; v < VOCAB; v += 256) orow[v] = row[v] - lse;
}

// ---------------------------------------------------------------------------
// Host launcher
// ---------------------------------------------------------------------------
static inline size_t align256(size_t x) { return (x + 255) & ~(size_t)255; }

static void launch_gemm(const half_t* A, int lda, const half_t* W, int ldw,
                        float* C, int ldc, const float* bias,
                        int M, int N, int K, hipStream_t stream) {
    dim3 grid((N + BROWS - 1) / BROWS, M / 32);
    gemm_f16_wmma<<<grid, 256, 0, stream>>>(A, lda, W, ldw, C, ldc, bias, N, K);
}

extern "C" void kernel_launch(void* const* d_in, const int* in_sizes, int n_in,
                              void* d_out, int out_size, void* d_ws, size_t ws_size,
                              hipStream_t stream) {
    const float* enc_out = (const float*)d_in[0];   // (B, L, HID)
    const float* enc_hid = (const float*)d_in[1];   // (1, B, HID)
    const int*   targets = (const int*)d_in[2];     // (B, MAXLEN)
    const float* embedding = (const float*)d_in[3]; // (VOCAB, WORD)
    const float* attn_W  = (const float*)d_in[4];   // (2*HID, HID)
    const float* attn_b  = (const float*)d_in[5];   // (HID,)
    const float* attn_v  = (const float*)d_in[6];   // (HID,)
    const float* W_ih    = (const float*)d_in[7];   // (3*HID, HID+WORD)
    const float* b_ih    = (const float*)d_in[8];   // (3*HID,)
    const float* W_hh    = (const float*)d_in[9];   // (3*HID, HID)
    const float* b_hh    = (const float*)d_in[10];  // (3*HID,)
    const float* out_W   = (const float*)d_in[11];  // (VOCAB, HID)
    const float* out_b   = (const float*)d_in[12];  // (VOCAB,)
    float* out = (float*)d_out;                     // (B, TSTEPS, VOCAB)

    char* p = (char*)d_ws;
    size_t off = 0;
    auto carve = [&](size_t bytes) -> void* {
        void* r = p + off;
        off += align256(bytes);
        return r;
    };
    half_t* enc16   = (half_t*)carve((size_t)BATCH * LENC * HID * 2);
    half_t* WhT16   = (half_t*)carve((size_t)HID * HID * 2);
    half_t* WeT16   = (half_t*)carve((size_t)HID * HID * 2);
    half_t* Wih16   = (half_t*)carve((size_t)3 * HID * (HID + WORD) * 2);
    half_t* Whh16   = (half_t*)carve((size_t)3 * HID * HID * 2);
    half_t* outW16  = (half_t*)carve((size_t)VOCAB * HID * 2);
    half_t* emb16   = (half_t*)carve((size_t)BATCH * TSTEPS * WORD * 2);
    float*  encU    = (float*)carve((size_t)BATCH * LENC * HID * 4);
    float*  Wh      = (float*)carve((size_t)BATCH * HID * 4);
    half_t* x16     = (half_t*)carve((size_t)BATCH * (HID + WORD) * 2);
    float*  gi      = (float*)carve((size_t)BATCH * 3 * HID * 4);
    float*  gh      = (float*)carve((size_t)BATCH * 3 * HID * 4);
    float*  h32     = (float*)carve((size_t)BATCH * HID * 4);
    half_t* h16     = (half_t*)carve((size_t)BATCH * HID * 2);
    float*  logits  = (float*)carve((size_t)BATCH * VOCAB * 4);
    (void)ws_size; (void)in_sizes; (void)n_in; (void)out_size;

    // ---- one-time prep ----
    attnw_transpose_kernel<<<(HID * HID) / 256, 256, 0, stream>>>(attn_W, WhT16, WeT16);
    {
        int n;
        n = 3 * HID * (HID + WORD);
        f32_to_f16_kernel<<<(n + 255) / 256, 256, 0, stream>>>(W_ih, Wih16, n);
        n = 3 * HID * HID;
        f32_to_f16_kernel<<<(n + 255) / 256, 256, 0, stream>>>(W_hh, Whh16, n);
        n = VOCAB * HID;
        f32_to_f16_kernel<<<(n + 255) / 256, 256, 0, stream>>>(out_W, outW16, n);
        n = BATCH * LENC * HID;
        f32_to_f16_kernel<<<(n + 255) / 256, 256, 0, stream>>>(enc_out, enc16, n);
    }
    embed_gather_kernel<<<(BATCH * TSTEPS * WORD) / 256, 256, 0, stream>>>(
        targets, embedding, emb16);
    h_init_kernel<<<(BATCH * HID) / 256, 256, 0, stream>>>(enc_hid, h32, h16);

    // encU = enc @ attn_W[HID:] + attn_b   (M=9216, N=1024, K=1024)
    launch_gemm(enc16, HID, WeT16, HID, encU, HID, attn_b,
                BATCH * LENC, HID, HID, stream);

    // ---- decode loop ----
    for (int t = 0; t < TSTEPS; ++t) {
        launch_gemm(h16, HID, WhT16, HID, Wh, HID, nullptr,
                    BATCH, HID, HID, stream);
        attn_ctx_kernel<<<BATCH, 256, 0, stream>>>(Wh, encU, enc_out, attn_v, x16);
        x_emb_copy_kernel<<<(BATCH * WORD) / 256, 256, 0, stream>>>(emb16, x16, t);
        launch_gemm(x16, HID + WORD, Wih16, HID + WORD, gi, 3 * HID, b_ih,
                    BATCH, 3 * HID, HID + WORD, stream);
        launch_gemm(h16, HID, Whh16, HID, gh, 3 * HID, b_hh,
                    BATCH, 3 * HID, HID, stream);
        gru_update_kernel<<<(BATCH * HID) / 256, 256, 0, stream>>>(gi, gh, h32, h16);
        launch_gemm(h16, HID, outW16, HID, logits, VOCAB, out_b,
                    BATCH, VOCAB, HID, stream);
        log_softmax_kernel<<<BATCH, 256, 0, stream>>>(logits, out, t);
    }
}
